// VGG16RoIHead_52836687675625
// MI455X (gfx1250) — matrix operand order repaired and checked by reading, not compiled, not run on hardware
//
#include <hip/hip_runtime.h>
#include <stdint.h>

typedef __attribute__((ext_vector_type(16))) __bf16 v16bf;
typedef __attribute__((ext_vector_type(8)))  float  v8f;

#if __has_builtin(__builtin_amdgcn_tensor_load_to_lds)
#define HAVE_TDM 1
typedef unsigned int v4u __attribute__((ext_vector_type(4)));
typedef int          v4i __attribute__((ext_vector_type(4)));
typedef int          v8i __attribute__((ext_vector_type(8)));
#else
#define HAVE_TDM 0
#endif

#define C_   512
#define H_   50
#define W_   50
#define HW_  2500
#define POOLN 7
#define NROI 512
#define FCK  25088   // 512*7*7
#define FCN  4096
#define BSTRIDE 40   // padded u16 stride for col-major B tile in LDS

__device__ __forceinline__ uint16_t f2bf(float f) {
  uint32_t u = __float_as_uint(f);
  u += 0x7FFFu + ((u >> 16) & 1u);   // round-to-nearest-even
  return (uint16_t)(u >> 16);
}

// ---------------- Kernel 1: per-pixel L2 norm over channels + rescale -------
__global__ void norm_kernel(const float* __restrict__ x,
                            const float* __restrict__ rescale,
                            float* __restrict__ xn) {
  int p = blockIdx.x * blockDim.x + threadIdx.x;
  if (p >= HW_) return;
  float s = 0.f;
  for (int c = 0; c < C_; ++c) { float v = x[c * HW_ + p]; s += v * v; }
  float inv = 1.f / sqrtf(s);
  for (int c = 0; c < C_; ++c) {
    xn[c * HW_ + p] = x[c * HW_ + p] * inv * rescale[c];
  }
}

// ---------------- Kernel 2: RoIPool (max), writes bf16 [NROI, 25088] --------
__global__ void roipool_kernel(const float* __restrict__ xn,
                               const float* __restrict__ rois,
                               uint16_t* __restrict__ pool) {
  int n = blockIdx.x;
  int x1 = (int)rintf(rois[n * 4 + 0] * 0.0625f);
  int y1 = (int)rintf(rois[n * 4 + 1] * 0.0625f);
  int x2 = (int)rintf(rois[n * 4 + 2] * 0.0625f);
  int y2 = (int)rintf(rois[n * 4 + 3] * 0.0625f);
  float roi_w = (float)max(x2 - x1 + 1, 1);
  float roi_h = (float)max(y2 - y1 + 1, 1);
  float bin_h = roi_h / (float)POOLN;
  float bin_w = roi_w / (float)POOLN;
  for (int o = threadIdx.x; o < FCK; o += blockDim.x) {
    int c  = o / 49;
    int r  = o % 49;
    int ph = r / 7, pw = r % 7;
    int hs = min(max((int)floorf((float)ph * bin_h) + y1, 0), H_);
    int he = min(max((int)ceilf((float)(ph + 1) * bin_h) + y1, 0), H_);
    int wsx = min(max((int)floorf((float)pw * bin_w) + x1, 0), W_);
    int wex = min(max((int)ceilf((float)(pw + 1) * bin_w) + x1, 0), W_);
    float m = 0.f;
    if (he > hs && wex > wsx) {
      m = -3.0e38f;
      for (int h = hs; h < he; ++h)
        for (int w = wsx; w < wex; ++w)
          m = fmaxf(m, xn[c * HW_ + h * W_ + w]);
    }
    pool[(size_t)n * FCK + o] = f2bf(m);
  }
}

// ---------------- Kernel 3: bf16 WMMA GEMM  out = act(A @ W + b) ------------
// A: bf16 [M,K] row-major.  W: fp32 [K,N] row-major (converted to bf16 while
// filling LDS).  Block tile 128(M)x128(N), 8 waves (4x2), wave tile 32x64 =
// 8x v_wmma_f32_16x16x32_bf16 per 32-wide K step.  LDS double buffered.
// A tile moved by the Tensor Data Mover (tensor_load_to_lds) issued by wave 0;
// B tile converted in-flight and stored K-contiguous per column so fragments
// load with ds_load_b128.
template <bool RELU, bool OUT_BF16, bool NCHECK>
__launch_bounds__(256)
__global__ void gemm_wmma(const uint16_t* __restrict__ A,
                          const float* __restrict__ Wt,
                          const float* __restrict__ bias,
                          void* __restrict__ outp,
                          int M, int N, int K) {
  __shared__ uint16_t lA[2][128 * 32];            // row-major [row][k]
  __shared__ uint16_t lB[2][128 * BSTRIDE];       // col-major [col][k], padded
  const int tid  = threadIdx.x;
  const int lane = tid & 31;
  const int wave = tid >> 5;
  const int wm   = wave & 3;   // 4 waves along M (32 rows each)
  const int wn   = wave >> 2;  // 2 waves along N (64 cols each)
  const int blockN = blockIdx.x * 128;
  const int blockM = blockIdx.y * 128;

  union Frag { uint32_t u[8]; uint4 q[2]; v16bf v; };

  v8f acc[2][4];
#pragma unroll
  for (int i = 0; i < 2; ++i)
#pragma unroll
    for (int j = 0; j < 4; ++j) acc[i][j] = {};

  const int KT = K >> 5;

  auto load_tiles = [&](int kt, int buf) {
    const int k0 = kt << 5;
#if HAVE_TDM
    // ---- A tile via Tensor Data Mover: 128 rows x 32 bf16, issued by wave 0.
    if (wave == 0) {
      uint64_t gaddr = (uint64_t)(uintptr_t)(A + (size_t)blockM * K + k0);
      uint32_t laddr = (uint32_t)(uintptr_t)&lA[buf][0];  // low 32 bits = LDS byte addr
      v4u g0;
      g0.x = 1u;                                  // count=1, user descriptor
      g0.y = laddr;                               // lds_addr
      g0.z = (uint32_t)gaddr;                     // global_addr[31:0]
      g0.w = ((uint32_t)(gaddr >> 32) & 0x01FFFFFFu) | (2u << 30);  // addr[56:32] | type=2
      v8i g1;
      g1.s0 = (int)(1u << 16);                              // data_size=1 (2 bytes)
      g1.s1 = (int)(((uint32_t)K & 0xFFFFu) << 16);         // tensor_dim0[15:0]
      g1.s2 = (int)((((uint32_t)K >> 16) & 0xFFFFu) |
                    (((uint32_t)M & 0xFFFFu) << 16));       // dim0[31:16] | dim1[15:0]
      g1.s3 = (int)((((uint32_t)M >> 16) & 0xFFFFu) |
                    (32u << 16));                           // dim1[31:16] | tile_dim0=32
      g1.s4 = (int)128u;                                    // tile_dim1=128, tile_dim2=0
      g1.s5 = (int)(uint32_t)K;                             // tensor_dim0_stride[31:0]
      g1.s6 = 0;                                            // stride[47:32] | dim1_stride lo
      g1.s7 = 0;
      v4i gz4 = {0, 0, 0, 0};
      v8i gz8 = {0, 0, 0, 0, 0, 0, 0, 0};
      __builtin_amdgcn_tensor_load_to_lds(g0, g1, gz4, gz4, gz8, 0);
    }
#else
    // ---- A tile fallback: 512 16B chunks / 256 threads = 2 each
#pragma unroll
    for (int it = 0; it < 2; ++it) {
      int idx = tid + it * 256;
      int row = idx >> 2;
      int seg = idx & 3;
      const uint4* src =
          reinterpret_cast<const uint4*>(A + (size_t)(blockM + row) * K + k0 + seg * 8);
      *reinterpret_cast<uint4*>(&lA[buf][row * 32 + seg * 8]) = *src;
    }
#endif
    // ---- B tile: 32(K) x 128(N), fp32 -> bf16, stored col-major [col][k]
    {
      int row = tid >> 3;  // k: 0..31
#pragma unroll
      for (int it = 0; it < 2; ++it) {
        int chunk = (tid & 7) + it * 8;   // 16 chunks of 8 cols
        int col0  = blockN + chunk * 8;
        uint16_t tmp[8];
        if (!NCHECK) {
          const float4* s0 =
              reinterpret_cast<const float4*>(Wt + (size_t)(k0 + row) * N + col0);
          float4 f0 = s0[0];
          float4 f1 = s0[1];
          tmp[0] = f2bf(f0.x); tmp[1] = f2bf(f0.y); tmp[2] = f2bf(f0.z); tmp[3] = f2bf(f0.w);
          tmp[4] = f2bf(f1.x); tmp[5] = f2bf(f1.y); tmp[6] = f2bf(f1.z); tmp[7] = f2bf(f1.w);
        } else {
#pragma unroll
          for (int e = 0; e < 8; ++e) {
            int col = col0 + e;
            float f = (col < N) ? Wt[(size_t)(k0 + row) * N + col] : 0.f;
            tmp[e] = f2bf(f);
          }
        }
#pragma unroll
        for (int e = 0; e < 8; ++e)
          lB[buf][(chunk * 8 + e) * BSTRIDE + row] = tmp[e];
      }
    }
  };

  auto tile_fence = [&]() {
#if HAVE_TDM
    __builtin_amdgcn_s_wait_tensorcnt(0);  // free for waves with TENSORcnt==0
#endif
    __syncthreads();
  };

  load_tiles(0, 0);
  tile_fence();

  for (int kt = 0; kt < KT; ++kt) {
    const int cur = kt & 1;
    if (kt + 1 < KT) load_tiles(kt + 1, cur ^ 1);

    const int half = lane >> 4;
    const int l15  = lane & 15;

    // A fragments: 16x32. lanes 0-15: elems 0-7 -> K0..7, 8-15 -> K16..23;
    // lanes 16-31: elems 0-7 -> K8..15, 8-15 -> K24..31  (pairs = u32 reads)
    Frag afr[2];
#pragma unroll
    for (int i = 0; i < 2; ++i) {
      int row = wm * 32 + i * 16 + l15;
      const uint32_t* r32 = reinterpret_cast<const uint32_t*>(&lA[cur][row * 32]);
#pragma unroll
      for (int j = 0; j < 4; ++j) {
        afr[i].u[j]     = r32[j + half * 4];
        afr[i].u[4 + j] = r32[8 + j + half * 4];
      }
    }
    // B fragments: col-major LDS -> K 8h..8h+7 and 16+8h..16+8h+7 per lane = 2x ds_load_b128
    Frag bfr[4];
#pragma unroll
    for (int j = 0; j < 4; ++j) {
      int col = wn * 64 + j * 16 + l15;
      const uint4* bp =
          reinterpret_cast<const uint4*>(&lB[cur][col * BSTRIDE + 8 * half]);
      bfr[j].q[0] = bp[0];
      bfr[j].q[1] = bp[2];
    }

#pragma unroll
    for (int i = 0; i < 2; ++i)
#pragma unroll
      for (int j = 0; j < 4; ++j)
        acc[i][j] = __builtin_amdgcn_wmma_f32_16x16x32_bf16(
            false, afr[i].v, false, bfr[j].v, (short)0, acc[i][j], false, false);

    tile_fence();
  }

  // Epilogue: D layout -> row = r + 8*(lane>=16), col = lane&15
  const int half = lane >> 4;
  const int l15  = lane & 15;
#pragma unroll
  for (int i = 0; i < 2; ++i) {
#pragma unroll
    for (int j = 0; j < 4; ++j) {
      int col = blockN + wn * 64 + j * 16 + l15;
      if (NCHECK && col >= N) continue;
      float bv = bias[col];
#pragma unroll
      for (int r = 0; r < 8; ++r) {
        int row = blockM + wm * 32 + i * 16 + r + 8 * half;
        float v = acc[i][j][r] + bv;
        if (RELU) v = fmaxf(v, 0.f);
        if (OUT_BF16)
          reinterpret_cast<uint16_t*>(outp)[(size_t)row * N + col] = f2bf(v);
        else
          reinterpret_cast<float*>(outp)[(size_t)row * N + col] = v;
      }
    }
  }
}

// ---------------- launcher ---------------------------------------------------
extern "C" void kernel_launch(void* const* d_in, const int* in_sizes, int n_in,
                              void* d_out, int out_size, void* d_ws, size_t ws_size,
                              hipStream_t stream) {
  const float* x       = (const float*)d_in[0];
  const float* rois    = (const float*)d_in[1];
  const float* rescale = (const float*)d_in[2];
  const float* W1      = (const float*)d_in[3];
  const float* b1      = (const float*)d_in[4];
  const float* W2      = (const float*)d_in[5];
  const float* b2      = (const float*)d_in[6];
  const float* Wloc    = (const float*)d_in[7];
  const float* bloc    = (const float*)d_in[8];
  const float* Wsc     = (const float*)d_in[9];
  const float* bsc     = (const float*)d_in[10];
  float* out = (float*)d_out;

  // workspace layout (bytes), ~39.2 MB total
  char* ws = (char*)d_ws;
  float*    xn  = (float*)(ws);                                   //  5,120,000 B
  uint16_t* pl  = (uint16_t*)(ws + 5120000);                      // 25,690,112 B
  uint16_t* h1  = (uint16_t*)(ws + 5120000 + 25690112);           //  4,194,304 B
  uint16_t* h2  = (uint16_t*)(ws + 5120000 + 25690112 + 4194304); //  4,194,304 B

  norm_kernel<<<(HW_ + 255) / 256, 256, 0, stream>>>(x, rescale, xn);
  roipool_kernel<<<NROI, 256, 0, stream>>>(xn, rois, pl);

  // FC1: [512,25088] @ [25088,4096] + relu -> bf16
  gemm_wmma<true, true, false><<<dim3(FCN / 128, NROI / 128), 256, 0, stream>>>(
      pl, W1, b1, h1, NROI, FCN, FCK);
  // FC2: [512,4096] @ [4096,4096] + relu -> bf16
  gemm_wmma<true, true, false><<<dim3(FCN / 128, NROI / 128), 256, 0, stream>>>(
      h1, W2, b2, h2, NROI, FCN, FCN);
  // heads -> fp32 into d_out (locs first, then scores)
  gemm_wmma<false, false, true><<<dim3(1, NROI / 128), 256, 0, stream>>>(
      h2, Wloc, bloc, out, NROI, 84, FCN);
  gemm_wmma<false, false, true><<<dim3(1, NROI / 128), 256, 0, stream>>>(
      h2, Wsc, bsc, out + (size_t)NROI * 84, NROI, 21, FCN);
}
